// DecoderLayer_79551384257150
// MI455X (gfx1250) — compile-verified, matrix operand order checked by reading, not run on hardware
//
#include <hip/hip_runtime.h>
#include <hip/hip_bf16.h>

// ---------------- problem constants ----------------
constexpr int Bn   = 2;
constexpr int Tn   = 2048;
constexpr int Cn   = 1024;
constexpr int Hn   = 16;
constexpr int HSn  = 64;
constexpr int Mn   = Bn * Tn;        // 4096 token rows
constexpr int HIDn = 2730;
constexpr int HIDP = 2752;           // HID padded to multiple of 64
constexpr int N12  = 2 * HIDP;       // 5504, W1|W2 concat (padded)
constexpr int NQKV = 3 * Cn;         // 3072

// ---------------- WMMA types ----------------
typedef __attribute__((ext_vector_type(16))) __bf16 v16bf;
typedef __attribute__((ext_vector_type(8)))  float  v8f;

__device__ __forceinline__ v8f v8f_zero() {
  v8f z = {0.f, 0.f, 0.f, 0.f, 0.f, 0.f, 0.f, 0.f};
  return z;
}

__device__ __forceinline__ unsigned short f2bf(float f) {
  unsigned int u = __float_as_uint(f);
  u += 0x7FFFu + ((u >> 16) & 1u);     // round-to-nearest-even
  return (unsigned short)(u >> 16);
}

__device__ __forceinline__ v8f wmma_bf16(v16bf a, v16bf b, v8f c) {
  return __builtin_amdgcn_wmma_f32_16x16x32_bf16(
      /*neg_a=*/false, a, /*neg_b=*/false, b,
      /*c_mod=*/(short)0, c, /*reuse_a=*/false, /*reuse_b=*/false);
}

// Load a 16x32 bf16 fragment (A layout; also valid as B frag when the tile is
// stored [n][k]).  Per ISA 7.12.2: lanes 0-15 -> row=lane, K 0..7 & 16..23;
// lanes 16-31 -> row=lane-16, K 8..15 & 24..31.  Two contiguous 16B loads.
__device__ __forceinline__ v16bf load_frag(const unsigned short* p, int ld) {
  int lane = threadIdx.x & 31;
  const unsigned short* q = p + (size_t)(lane & 15) * ld + ((lane >> 4) << 3);
  union { v16bf v; uint4 u[2]; } r;
  r.u[0] = *(const uint4*)(q);
  r.u[1] = *(const uint4*)(q + 16);
  return r.v;
}

// ---------------- LayerNorm -> bf16 ----------------
__global__ __launch_bounds__(256)
void ln_bf16_kernel(const float* __restrict__ x, const float* __restrict__ gamma,
                    const float* __restrict__ beta, unsigned short* __restrict__ out) {
  int row = blockIdx.x;
  const float* xr = x + (size_t)row * Cn;
  __shared__ float red[256];
  float s = 0.f, s2 = 0.f;
  for (int i = threadIdx.x; i < Cn; i += 256) { float v = xr[i]; s += v; s2 += v * v; }
  red[threadIdx.x] = s; __syncthreads();
  for (int st = 128; st > 0; st >>= 1) {
    if ((int)threadIdx.x < st) red[threadIdx.x] += red[threadIdx.x + st];
    __syncthreads();
  }
  float mean = red[0] / (float)Cn;
  __syncthreads();
  red[threadIdx.x] = s2; __syncthreads();
  for (int st = 128; st > 0; st >>= 1) {
    if ((int)threadIdx.x < st) red[threadIdx.x] += red[threadIdx.x + st];
    __syncthreads();
  }
  float var = red[0] / (float)Cn - mean * mean;
  float rstd = rsqrtf(var + 1e-3f);
  unsigned short* orow = out + (size_t)row * Cn;
  for (int i = threadIdx.x; i < Cn; i += 256)
    orow[i] = f2bf((xr[i] - mean) * rstd * gamma[i] + beta[i]);
}

// ---------------- weight conversions ----------------
__global__ void conv_wqkv(const float* __restrict__ Wq, const float* __restrict__ Wk,
                          const float* __restrict__ Wv, unsigned short* __restrict__ out) {
  int idx = blockIdx.x * 256 + threadIdx.x;          // over Cn*NQKV
  if (idx >= Cn * NQKV) return;
  int c = idx / NQKV, n = idx - c * NQKV;
  int which = n >> 10, nn = n & 1023;
  int h = nn >> 6, d = nn & 63;
  const float* W = (which == 0) ? Wq : (which == 1) ? Wk : Wv;
  out[idx] = f2bf(W[((size_t)h * Cn + c) * HSn + d]);   // (H,C,HS) -> (C, 3C)
}

__global__ void conv_f32_bf16(const float* __restrict__ in, unsigned short* __restrict__ out, int n) {
  int idx = blockIdx.x * 256 + threadIdx.x;
  if (idx < n) out[idx] = f2bf(in[idx]);
}

__global__ void conv_w12(const float* __restrict__ W1, const float* __restrict__ W2,
                         unsigned short* __restrict__ out) {
  int idx = blockIdx.x * 256 + threadIdx.x;          // over Cn*N12
  if (idx >= Cn * N12) return;
  int c = idx / N12, j = idx - c * N12;
  float v = 0.f;
  if (j < HIDn)                          v = W1[(size_t)c * HIDn + j];
  else if (j >= HIDP && j < HIDP + HIDn) v = W2[(size_t)c * HIDn + (j - HIDP)];
  out[idx] = f2bf(v);
}

__global__ void conv_w3p(const float* __restrict__ W3, unsigned short* __restrict__ out) {
  int idx = blockIdx.x * 256 + threadIdx.x;          // over HIDP*Cn
  if (idx >= HIDP * Cn) return;
  int r = idx >> 10, c = idx & 1023;
  float v = (r < HIDn) ? W3[(size_t)r * Cn + c] : 0.f;
  out[idx] = f2bf(v);
}

// ---------------- QKV split: (M,3C) f32 -> per-head [b,h,t,d] bf16 ----------------
__global__ void split_qkv(const float* __restrict__ qkv, unsigned short* __restrict__ qb,
                          unsigned short* __restrict__ kb, unsigned short* __restrict__ vb) {
  int idx = blockIdx.x * 256 + threadIdx.x;          // over Mn*Cn
  if (idx >= Mn * Cn) return;
  int bt = idx >> 10, n = idx & 1023;
  int h = n >> 6, d = n & 63;
  int b = bt >> 11, t = bt & 2047;
  const float* base = qkv + (size_t)bt * NQKV;
  size_t o = (((size_t)(b * Hn + h)) * Tn + t) * HSn + d;
  qb[o] = f2bf(base[n]);
  kb[o] = f2bf(base[Cn + n]);
  vb[o] = f2bf(base[2 * Cn + n]);
}

// ---------------- bf16 WMMA GEMM: C = A(MxK) * B(KxN) [+R] ----------------
// Block tile 128x128xBK64, 256 threads = 8 waves (4 in M, 2 in N).
// Wave tile 32x64 -> 8 f32 accumulators, A frags reused 4x, B frags 2x:
// 16 WMMAs per stage vs 24 b128 fragment loads, barriers halved vs BK=32.
template <int EPI>   // 0: none, 1: add residual R
__global__ __launch_bounds__(256)
void gemm_bf16(const unsigned short* __restrict__ A, const unsigned short* __restrict__ Bm,
               float* __restrict__ Cm, const float* __restrict__ R,
               int Mdim, int Ndim, int Kdim) {
  constexpr int BK = 64;
  __shared__ __align__(16) unsigned short As[128 * BK];   // [m][k]
  __shared__ __align__(16) unsigned short Bs[128 * BK];   // [n][k] (transposed stage)
  int m0 = blockIdx.y * 128;
  int n0 = blockIdx.x * 128;
  int tid = threadIdx.x;
  int wave = tid >> 5, lane = tid & 31;
  int wm = wave >> 1, wn = wave & 1;                      // 4 x 2 wave grid
  int col = lane & 15, hlf = lane >> 4, rbase = hlf * 8;

  v8f acc0[4], acc1[4];
#pragma unroll
  for (int n = 0; n < 4; ++n) { acc0[n] = v8f_zero(); acc1[n] = v8f_zero(); }

  for (int k0 = 0; k0 < Kdim; k0 += BK) {
    __syncthreads();
#pragma unroll
    for (int j = 0; j < 4; ++j) {                 // stage A: 128x64, 4 x 16B/thread
      int u = tid * 4 + j;
      int arow = u >> 3, acol8 = (u & 7) << 3;
      *(uint4*)(&As[arow * BK + acol8]) =
          *(const uint4*)(&A[(size_t)(m0 + arow) * Kdim + k0 + acol8]);
      if (k0 + BK < Kdim)                          // gfx1250 global_prefetch_b8
        __builtin_prefetch(&A[(size_t)(m0 + arow) * Kdim + k0 + BK + acol8], 0, 3);
    }
#pragma unroll
    for (int j = 0; j < 4; ++j) {                 // stage B: 64x128 -> [n][k]
      int u = tid * 4 + j;
      int bkrow = u >> 4, bn8 = (u & 15) << 3;
      uint4 bv = *(const uint4*)(&Bm[(size_t)(k0 + bkrow) * Ndim + n0 + bn8]);
      const unsigned short* bs = (const unsigned short*)&bv;
#pragma unroll
      for (int i = 0; i < 8; ++i) Bs[(bn8 + i) * BK + bkrow] = bs[i];
      if (k0 + BK < Kdim)
        __builtin_prefetch(&Bm[(size_t)(k0 + BK + bkrow) * Ndim + n0 + bn8], 0, 3);
    }
    __syncthreads();

#pragma unroll
    for (int kk = 0; kk < BK; kk += 32) {
      v16bf af0 = load_frag(&As[(wm * 32) * BK + kk], BK);
      v16bf af1 = load_frag(&As[(wm * 32 + 16) * BK + kk], BK);
#pragma unroll
      for (int n = 0; n < 4; ++n) {
        v16bf bfn = load_frag(&Bs[(wn * 64 + n * 16) * BK + kk], BK);
        acc0[n] = wmma_bf16(af0, bfn, acc0[n]);
        acc1[n] = wmma_bf16(af1, bfn, acc1[n]);
      }
    }
  }

  int crow = m0 + wm * 32 + rbase;
#pragma unroll
  for (int n = 0; n < 4; ++n) {
    int ccol = n0 + wn * 64 + n * 16 + col;
#pragma unroll
    for (int r = 0; r < 8; ++r) {
      size_t i0 = (size_t)(crow + r) * Ndim + ccol;
      size_t i1 = (size_t)(crow + 16 + r) * Ndim + ccol;
      float v0 = acc0[n][r], v1 = acc1[n][r];
      if (EPI == 1) { v0 += R[i0]; v1 += R[i1]; }
      Cm[i0] = v0;
      Cm[i1] = v1;
    }
  }
}

// ---------------- causal flash attention (bf16 WMMA, f32 online softmax) ----------
// Grid: (T/128, B*H).  256 threads = 8 waves; wave owns 16 query rows.
__global__ __launch_bounds__(256)
void attn_kernel(const unsigned short* __restrict__ qb, const unsigned short* __restrict__ kbuf,
                 const unsigned short* __restrict__ vbuf, unsigned short* __restrict__ ctx) {
  int t0 = blockIdx.x * 128;
  int bh = blockIdx.y;
  int b = bh >> 4, h = bh & 15;
  int tid = threadIdx.x, wave = tid >> 5, lane = tid & 31;
  int col = lane & 15, hlf = lane >> 4, rbase = hlf * 8;
  int qrow0 = t0 + wave * 16;

  __shared__ __align__(16) unsigned short Kt[32 * 64];      // [s][d]
  __shared__ __align__(16) unsigned short Vt[64 * 32];      // [d][s]
  __shared__ __align__(16) unsigned short Pl[8][16 * 32];   // per-wave P tile

  const size_t headoff = (size_t)bh * Tn * HSn;

  // Q fragments for this wave's 16 rows (held in registers for the whole pass)
  v16bf qf0 = load_frag(qb + headoff + (size_t)qrow0 * 64, 64);        // d 0..31
  v16bf qf1 = load_frag(qb + headoff + (size_t)qrow0 * 64 + 32, 64);   // d 32..63

  v8f o0 = v8f_zero(), o1 = v8f_zero(), o2 = v8f_zero(), o3 = v8f_zero();
  float m[8], l[8];
#pragma unroll
  for (int r = 0; r < 8; ++r) { m[r] = -1e30f; l[r] = 0.f; }

  int nkb = (t0 + 128) >> 5;                // causal: keys up to block's last row
  for (int kb = 0; kb < nkb; ++kb) {
    int s0 = kb << 5;
    __syncthreads();
    {   // stage 32 keys: K row-major [s][d], V transposed [d][s]
      int srow = tid >> 3, d8 = (tid & 7) << 3;
      *(uint4*)(&Kt[srow * 64 + d8]) =
          *(const uint4*)(&kbuf[headoff + (size_t)(s0 + srow) * 64 + d8]);
      uint4 vv = *(const uint4*)(&vbuf[headoff + (size_t)(s0 + srow) * 64 + d8]);
      const unsigned short* vs = (const unsigned short*)&vv;
#pragma unroll
      for (int i = 0; i < 8; ++i) Vt[(d8 + i) * 32 + srow] = vs[i];
    }
    __syncthreads();

    if (s0 <= qrow0 + 15) {                 // wave-uniform causal skip
      v8f sa0 = v8f_zero(), sa1 = v8f_zero();
#pragma unroll
      for (int dc = 0; dc < 2; ++dc) {
        v16bf qf = dc ? qf1 : qf0;
        v16bf kf0 = load_frag(&Kt[0 * 64 + dc * 32], 64);     // keys s0..s0+15
        v16bf kf1 = load_frag(&Kt[16 * 64 + dc * 32], 64);    // keys s0+16..s0+31
        sa0 = wmma_bf16(qf, kf0, sa0);
        sa1 = wmma_bf16(qf, kf1, sa1);
      }
      float p0[8], p1[8], alpha[8];
#pragma unroll
      for (int r = 0; r < 8; ++r) {
        int tg = qrow0 + rbase + r;
        float v0 = sa0[r] * 0.125f;                    // 1/sqrt(64)
        float v1 = sa1[r] * 0.125f;
        if (s0 + col > tg)      v0 = -1e30f;           // causal mask
        if (s0 + 16 + col > tg) v1 = -1e30f;
        float mx = fmaxf(v0, v1);
#pragma unroll
        for (int msk = 1; msk <= 8; msk <<= 1)
          mx = fmaxf(mx, __shfl_xor(mx, msk, 32));     // row lives in a 16-lane half
        mx = fmaxf(mx, m[r]);
        alpha[r] = __expf(m[r] - mx);
        m[r] = mx;
        p0[r] = __expf(v0 - mx);
        p1[r] = __expf(v1 - mx);
        float rs = p0[r] + p1[r];
#pragma unroll
        for (int msk = 1; msk <= 8; msk <<= 1)
          rs += __shfl_xor(rs, msk, 32);
        l[r] = l[r] * alpha[r] + rs;
      }
#pragma unroll
      for (int r = 0; r < 8; ++r) {
        o0[r] *= alpha[r]; o1[r] *= alpha[r]; o2[r] *= alpha[r]; o3[r] *= alpha[r];
        Pl[wave][(rbase + r) * 32 + col]      = f2bf(p0[r]);
        Pl[wave][(rbase + r) * 32 + 16 + col] = f2bf(p1[r]);
      }
      // intra-wave LDS store->load hazard: drain the CDNA5 DS counter
      asm volatile("s_wait_dscnt 0" ::: "memory");
      v16bf pf  = load_frag(&Pl[wave][0], 32);
      v16bf vf0 = load_frag(&Vt[0 * 32], 32);
      v16bf vf1 = load_frag(&Vt[16 * 32], 32);
      v16bf vf2 = load_frag(&Vt[32 * 32], 32);
      v16bf vf3 = load_frag(&Vt[48 * 32], 32);
      o0 = wmma_bf16(pf, vf0, o0);
      o1 = wmma_bf16(pf, vf1, o1);
      o2 = wmma_bf16(pf, vf2, o2);
      o3 = wmma_bf16(pf, vf3, o3);
    }
  }

#pragma unroll
  for (int r = 0; r < 8; ++r) {
    float inv = 1.0f / l[r];
    int tg = qrow0 + rbase + r;
    size_t oo = ((size_t)b * Tn + tg) * Cn + h * 64 + col;   // ctx: (B*T, C) bf16
    ctx[oo]      = f2bf(o0[r] * inv);
    ctx[oo + 16] = f2bf(o1[r] * inv);
    ctx[oo + 32] = f2bf(o2[r] * inv);
    ctx[oo + 48] = f2bf(o3[r] * inv);
  }
}

// ---------------- SwiGLU: g = silu(G1) * G2 -> bf16 ----------------
__global__ void swiglu_kernel(const float* __restrict__ g12, unsigned short* __restrict__ out) {
  int idx = blockIdx.x * 256 + threadIdx.x;          // over Mn*HIDP
  if (idx >= Mn * HIDP) return;
  int mrow = idx / HIDP, j = idx - mrow * HIDP;
  float g1 = g12[(size_t)mrow * N12 + j];
  float g2 = g12[(size_t)mrow * N12 + HIDP + j];
  float s = g1 / (1.f + __expf(-g1));
  out[idx] = f2bf(s * g2);
}

// ---------------- launch ----------------
extern "C" void kernel_launch(void* const* d_in, const int* in_sizes, int n_in,
                              void* d_out, int out_size, void* d_ws, size_t ws_size,
                              hipStream_t stream) {
  const float* x     = (const float*)d_in[0];
  const float* Wq    = (const float*)d_in[1];
  const float* Wk    = (const float*)d_in[2];
  const float* Wv    = (const float*)d_in[3];
  const float* Wo    = (const float*)d_in[4];
  const float* W1    = (const float*)d_in[5];
  const float* W2    = (const float*)d_in[6];
  const float* W3    = (const float*)d_in[7];
  const float* gamma = (const float*)d_in[8];
  const float* beta  = (const float*)d_in[9];

  char* ws = (char*)d_ws;
  size_t off = 0;
  auto take = [&](size_t bytes) -> void* {
    void* p = ws + off;
    off += (bytes + 255) & ~(size_t)255;
    return p;
  };
  unsigned short* hnb   = (unsigned short*)take((size_t)Mn * Cn * 2);
  unsigned short* wqkvb = (unsigned short*)take((size_t)Cn * NQKV * 2);
  float*          qkv   = (float*)take((size_t)Mn * NQKV * 4);
  unsigned short* qbuf  = (unsigned short*)take((size_t)Mn * Cn * 2);
  unsigned short* kbufp = (unsigned short*)take((size_t)Mn * Cn * 2);
  unsigned short* vbufp = (unsigned short*)take((size_t)Mn * Cn * 2);
  unsigned short* ctxb  = (unsigned short*)take((size_t)Mn * Cn * 2);
  unsigned short* wob   = (unsigned short*)take((size_t)Cn * Cn * 2);
  float*          x1    = (float*)take((size_t)Mn * Cn * 4);
  unsigned short* hn2b  = (unsigned short*)take((size_t)Mn * Cn * 2);
  unsigned short* w12b  = (unsigned short*)take((size_t)Cn * N12 * 2);
  float*          g12   = (float*)take((size_t)Mn * N12 * 4);
  unsigned short* gbuf  = (unsigned short*)take((size_t)Mn * HIDP * 2);
  unsigned short* w3b   = (unsigned short*)take((size_t)HIDP * Cn * 2);
  (void)ws_size; (void)in_sizes; (void)n_in; (void)out_size;

  auto cdiv = [](int a, int b) { return (a + b - 1) / b; };

  // 1) LN(x) -> bf16, weights -> bf16 (padded/fused layouts)
  ln_bf16_kernel<<<Mn, 256, 0, stream>>>(x, gamma, beta, hnb);
  conv_wqkv<<<cdiv(Cn * NQKV, 256), 256, 0, stream>>>(Wq, Wk, Wv, wqkvb);
  conv_f32_bf16<<<cdiv(Cn * Cn, 256), 256, 0, stream>>>(Wo, wob, Cn * Cn);
  conv_w12<<<cdiv(Cn * N12, 256), 256, 0, stream>>>(W1, W2, w12b);
  conv_w3p<<<cdiv(HIDP * Cn, 256), 256, 0, stream>>>(W3, w3b);

  // 2) QKV = hn @ Wqkv   (4096 x 3072 x 1024)
  gemm_bf16<0><<<dim3(NQKV / 128, Mn / 128), 256, 0, stream>>>(
      hnb, wqkvb, qkv, nullptr, Mn, NQKV, Cn);
  split_qkv<<<cdiv(Mn * Cn, 256), 256, 0, stream>>>(qkv, qbuf, kbufp, vbufp);

  // 3) causal flash attention -> ctx bf16 (B*T, C)
  attn_kernel<<<dim3(Tn / 128, Bn * Hn), 256, 0, stream>>>(qbuf, kbufp, vbufp, ctxb);

  // 4) x1 = x + ctx @ Wo
  gemm_bf16<1><<<dim3(Cn / 128, Mn / 128), 256, 0, stream>>>(
      ctxb, wob, x1, x, Mn, Cn, Cn);

  // 5) LN(x1) -> bf16 ; G12 = hn2 @ [W1|W2]   (4096 x 5504 x 1024)
  ln_bf16_kernel<<<Mn, 256, 0, stream>>>(x1, gamma, beta, hn2b);
  gemm_bf16<0><<<dim3(N12 / 128, Mn / 128), 256, 0, stream>>>(
      hn2b, w12b, g12, nullptr, Mn, N12, Cn);

  // 6) g = silu(G1)*G2 ; out = x1 + g @ W3p   (4096 x 1024 x 2752)
  swiglu_kernel<<<cdiv(Mn * HIDP, 256), 256, 0, stream>>>(g12, gbuf);
  gemm_bf16<1><<<dim3(Cn / 128, Mn / 128), 256, 0, stream>>>(
      gbuf, w3b, (float*)d_out, x1, Mn, Cn, HIDP);
}
